// DCNBlock_85237920956940
// MI455X (gfx1250) — compile-verified, matrix operand order checked by reading, not run on hardware
//
#include <hip/hip_runtime.h>

typedef __attribute__((ext_vector_type(16))) _Float16 v16h;
typedef __attribute__((ext_vector_type(8)))  float    v8f;

#define HH   256
#define WWID 256
#define HW   65536
#define CIN  64
#define OUTC 64
#define NPIX 131072          // B*H*W
#define KTOT 576             // CIN*9

// ---------------------------------------------------------------- cast main conv w -> f16 [64][576]
__global__ __launch_bounds__(256) void cast_w_kernel(const float* __restrict__ w,
                                                     _Float16* __restrict__ wh) {
  int i = blockIdx.x * 256 + threadIdx.x;
  if (i < OUTC * KTOT) wh[i] = (_Float16)w[i];
}

// ---------------------------------------------------------------- cast offset conv w -> f16 [32][576] (rows 27..31 zero)
__global__ __launch_bounds__(256) void cast_woff_kernel(const float* __restrict__ w_off,
                                                        _Float16* __restrict__ whoff) {
  int i = blockIdx.x * 256 + threadIdx.x;          // 32*576 = 18432
  if (i < 32 * KTOT) {
    int o = i / KTOT;
    whoff[i] = (o < 27) ? (_Float16)w_off[i] : (_Float16)0.0f;
  }
}

// ---------------------------------------------------------------- offset conv via WMMA im2col GEMM
// M = 16-pixel strip per wave, N = 32 (27 used), K = 576
__global__ __launch_bounds__(256) void offset_conv_wmma_kernel(
    const float* __restrict__ x, const _Float16* __restrict__ whoff,
    const float* __restrict__ b_off, float* __restrict__ omP) {
  const int lane  = threadIdx.x & 31;
  const int wave  = threadIdx.x >> 5;
  const int strip = blockIdx.x * 8 + wave;          // 8192 strips of 16 pixels
  const int p0    = strip * 16;
  const int mrow  = lane & 15;
  const int khalf = lane >> 4;
  const int p     = p0 + mrow;
  const int b     = p >> 16;
  const int hw    = p & (HW - 1);
  const int h     = hw >> 8;
  const int w0    = hw & 255;

  // per-tap clamped address offset + zero-pad validity
  int   ofs[9];
  float okf[9];
#pragma unroll
  for (int t = 0; t < 9; ++t) {
    int yy = h + t / 3 - 1, xx = w0 + t % 3 - 1;
    bool ok = (yy >= 0) && (yy < HH) && (xx >= 0) && (xx < WWID);
    int yc = min(max(yy, 0), HH - 1), xc = min(max(xx, 0), WWID - 1);
    ofs[t] = b * CIN * HW + yc * WWID + xc;
    okf[t] = ok ? 1.0f : 0.0f;
  }

  v8f acc[2] = {};
  for (int kb = 0; kb < KTOT; kb += 32) {
    v16h afrag;
#pragma unroll
    for (int e = 0; e < 16; ++e) {
      int Koff = (e < 8) ? (khalf * 8 + e) : (8 + khalf * 8 + e);
      int K = kb + Koff;
      int c = K / 9;
      int t = K - c * 9;
      afrag[e] = (_Float16)(x[c * HW + ofs[t]] * okf[t]);
    }
#pragma unroll
    for (int nt = 0; nt < 2; ++nt) {
      const _Float16* wrow = whoff + (nt * 16 + (lane & 15)) * KTOT + kb;
      v16h bfrag;
#pragma unroll
      for (int e = 0; e < 16; ++e) {
        int Koff = (e < 8) ? (khalf * 8 + e) : (8 + khalf * 8 + e);
        bfrag[e] = wrow[Koff];
      }
      acc[nt] = __builtin_amdgcn_wmma_f32_16x16x32_f16(
          false, afrag, false, bfrag, (short)0, acc[nt], false, false);
    }
  }

  // epilogue: bias, sigmoid on mask channels (18..26), store [b][27][hw]
#pragma unroll
  for (int nt = 0; nt < 2; ++nt) {
    int o = nt * 16 + (lane & 15);
    if (o < 27) {
      float bo = b_off[o];
#pragma unroll
      for (int r = 0; r < 8; ++r) {
        int pp = p0 + khalf * 8 + r;                // D layout: M = r + 8*khalf
        float v = acc[nt][r] + bo;
        if (o >= 18) v = 1.0f / (1.0f + __expf(-v));
        omP[((pp >> 16) * 27 + o) * HW + (pp & (HW - 1))] = v;
      }
    }
  }
}

// ---------------------------------------------------------------- fused sample + WMMA GEMM
__global__ __launch_bounds__(256) void dcn_gemm_kernel(
    const float* __restrict__ x, const float* __restrict__ omP,
    const _Float16* __restrict__ wh, const float* __restrict__ bias,
    float* __restrict__ pre, float* __restrict__ stats) {
  __shared__ float red[2 * OUTC];
  for (int i = threadIdx.x; i < 2 * OUTC; i += 256) red[i] = 0.0f;
  __syncthreads();

  const int lane  = threadIdx.x & 31;
  const int wave  = threadIdx.x >> 5;
  const int strip = blockIdx.x * 8 + wave;          // 8192 strips of 16 pixels
  const int p0    = strip * 16;
  const int mrow  = lane & 15;
  const int khalf = lane >> 4;                      // CDNA5 16-bit A/B layout half
  const int p     = p0 + mrow;
  const int b     = p >> 16;
  const int hw    = p & (HW - 1);
  const int h     = hw >> 8;
  const int w0    = hw & 255;

  // per-tap bilinear gather setup (mask folded into corner weights)
  int   ga[9][4];
  float gw[9][4];
  const float* om_b = omP + b * 27 * HW;
#pragma unroll
  for (int k = 0; k < 9; ++k) {
    float oy = om_b[k * HW + hw];
    float ox = om_b[(9 + k) * HW + hw];
    float m  = om_b[(18 + k) * HW + hw];
    float py = (float)(h + k / 3 - 1) + oy;
    float px = (float)(w0 + k % 3 - 1) + ox;
    float fy = floorf(py), fx = floorf(px);
    float wy = py - fy, wx = px - fx;
    int y0 = (int)fy, x0 = (int)fx;
    float cw[4] = {(1.0f - wy) * (1.0f - wx) * m, (1.0f - wy) * wx * m,
                   wy * (1.0f - wx) * m, wy * wx * m};
#pragma unroll
    for (int corner = 0; corner < 4; ++corner) {
      int yy = y0 + (corner >> 1), xx = x0 + (corner & 1);
      bool ok = (yy >= 0) && (yy < HH) && (xx >= 0) && (xx < WWID);
      int yc = min(max(yy, 0), HH - 1), xc = min(max(xx, 0), WWID - 1);
      ga[k][corner] = b * CIN * HW + yc * WWID + xc;
      gw[k][corner] = ok ? cw[corner] : 0.0f;
    }
  }

  v8f acc[4] = {};                                  // 4 N-tiles of 16 channels
  for (int kb = 0; kb < KTOT; kb += 32) {
    v16h afrag;
#pragma unroll
    for (int e = 0; e < 16; ++e) {
      int Koff = (e < 8) ? (khalf * 8 + e) : (8 + khalf * 8 + e);
      int K = kb + Koff;
      int c = K / 9;
      int k = K - c * 9;
      const float* xc = x + c * HW;
      float v = xc[ga[k][0]] * gw[k][0] + xc[ga[k][1]] * gw[k][1]
              + xc[ga[k][2]] * gw[k][2] + xc[ga[k][3]] * gw[k][3];
      afrag[e] = (_Float16)v;
    }
#pragma unroll
    for (int nt = 0; nt < 4; ++nt) {
      const _Float16* wrow = wh + (nt * 16 + (lane & 15)) * KTOT + kb;
      v16h bfrag;
#pragma unroll
      for (int e = 0; e < 16; ++e) {
        int Koff = (e < 8) ? (khalf * 8 + e) : (8 + khalf * 8 + e);
        bfrag[e] = wrow[Koff];
      }
      acc[nt] = __builtin_amdgcn_wmma_f32_16x16x32_f16(
          false, afrag, false, bfrag, (short)0, acc[nt], false, false);
    }
  }

  // epilogue: bias, store pre-BN, per-channel sum/sumsq via LDS atomics
#pragma unroll
  for (int nt = 0; nt < 4; ++nt) {
    int o = nt * 16 + (lane & 15);
    float bv = bias[o];
    float s = 0.0f, s2 = 0.0f;
#pragma unroll
    for (int r = 0; r < 8; ++r) {
      int pp = p0 + khalf * 8 + r;                  // D layout: M = r + 8*khalf
      float v = acc[nt][r] + bv;
      pre[((pp >> 16) * OUTC + o) * HW + (pp & (HW - 1))] = v;
      s += v;
      s2 += v * v;
    }
    atomicAdd(&red[o], s);
    atomicAdd(&red[OUTC + o], s2);
  }
  __syncthreads();
  for (int i = threadIdx.x; i < 2 * OUTC; i += 256) atomicAdd(&stats[i], red[i]);
}

// ---------------------------------------------------------------- BN finalize
__global__ void bn_finalize_kernel(const float* __restrict__ stats,
                                   const float* __restrict__ gamma,
                                   const float* __restrict__ beta,
                                   float* __restrict__ sc) {
  int o = threadIdx.x;
  if (o < OUTC) {
    float n   = (float)NPIX;
    float mu  = stats[o] / n;
    float var = stats[OUTC + o] / n - mu * mu;
    float s   = gamma[o] * rsqrtf(var + 1e-5f);
    sc[o]        = s;
    sc[OUTC + o] = beta[o] - mu * s;
  }
}

// ---------------------------------------------------------------- BN + ReLU + residual
__global__ __launch_bounds__(256) void bn_apply_kernel(
    const float* __restrict__ pre, const float* __restrict__ sc,
    const float* __restrict__ x, float* __restrict__ out) {
  int i = blockIdx.x * 256 + threadIdx.x;           // [b][o][hw], 8388608 total
  int o = (i >> 16) & 63;
  float v = pre[i] * sc[o] + sc[64 + o];
  out[i] = fmaxf(v, 0.0f) + x[i];
}

// ---------------------------------------------------------------- launch
extern "C" void kernel_launch(void* const* d_in, const int* in_sizes, int n_in,
                              void* d_out, int out_size, void* d_ws, size_t ws_size,
                              hipStream_t stream) {
  const float* x     = (const float*)d_in[0];
  const float* w_off = (const float*)d_in[1];
  const float* b_off = (const float*)d_in[2];
  const float* w     = (const float*)d_in[3];
  const float* bias  = (const float*)d_in[4];
  const float* gamma = (const float*)d_in[5];
  const float* beta  = (const float*)d_in[6];
  float* out = (float*)d_out;

  char* ws = (char*)d_ws;
  float*    omP   = (float*)(ws);                     // 2*27*65536 f32 = 14155776 B
  _Float16* wh    = (_Float16*)(ws + 14155776);       // 64*576 f16     =    73728 B
  _Float16* whoff = (_Float16*)(ws + 14229504);       // 32*576 f16     =    36864 B
  float*    pre   = (float*)(ws + 14266368);          // 2*64*65536 f32 = 33554432 B
  float*    stats = (float*)(ws + 47820800);          // 128 f32
  float*    sc    = stats + 128;                      // 128 f32

  hipMemsetAsync(stats, 0, 128 * sizeof(float), stream);
  cast_w_kernel<<<144, 256, 0, stream>>>(w, wh);
  cast_woff_kernel<<<72, 256, 0, stream>>>(w_off, whoff);
  offset_conv_wmma_kernel<<<1024, 256, 0, stream>>>(x, whoff, b_off, omP);
  dcn_gemm_kernel<<<1024, 256, 0, stream>>>(x, omP, wh, bias, pre, stats);
  bn_finalize_kernel<<<1, 64, 0, stream>>>(stats, gamma, beta, sc);
  bn_apply_kernel<<<32768, 256, 0, stream>>>(pre, sc, x, out);
}
